// Re2Alignment_57672820851281
// MI455X (gfx1250) — compile-verified, matrix operand order checked by reading, not run on hardware
//
#include <hip/hip_runtime.h>
#include <hip/hip_bf16.h>
#include <stddef.h>
#include <stdint.h>

// ---------------------------------------------------------------------------
// Re2 alignment on MI455X (gfx1250, wave32):
//   align_a = FlashAttn(Q=ha, K=hb, V=b, key_mask=mask_b, scale=tau)
//   align_b = FlashAttn(Q=hb, K=ha, V=a, key_mask=mask_a, scale=tau)
// with ha/hb = relu(x @ weight_norm(v,g)^T + bias), all matmuls on
// v_wmma_f32_16x16x32_bf16 (f32 accumulate), softmax in f32.
// K tiles are streamed into LDS by the Tensor Data Mover (tensor_load_to_lds,
// TENSORcnt-tracked) with 2-deep double buffering.
// ---------------------------------------------------------------------------

typedef __attribute__((ext_vector_type(16))) __bf16        v16bf;
typedef __attribute__((ext_vector_type(8)))  float         v8f;
typedef __attribute__((ext_vector_type(4)))  unsigned int  u32x4;
typedef __attribute__((ext_vector_type(8)))  int           i32x8;
typedef __attribute__((ext_vector_type(4)))  int           i32x4;

union ABOp { v16bf v; u32x4 q[2]; };

// Load one WMMA A/B operand half-pattern: elements 0..7 are 8 contiguous
// bf16 at `base`, elements 8..15 are 8 contiguous bf16 at `base + 16`.
// (ISA 7.12.2: 16-bit 16x32 operand, K = (e<8 ? e : e+8) + half*8.)
static __device__ __forceinline__ v16bf load_op16(const __bf16* base) {
  ABOp t;
  t.q[0] = *(const u32x4*)(base);
  t.q[1] = *(const u32x4*)(base + 16);
  return t.v;
}

static __device__ __forceinline__ v8f wmma_bf16(v16bf a, v16bf b, v8f c) {
  return __builtin_amdgcn_wmma_f32_16x16x32_bf16(false, a, false, b,
                                                 (short)0, c, false, false);
}

// ---- Tensor Data Mover: async 2D tile load Global -> LDS (TENSORcnt) ------
// D# per CDNA5 ISA ch.8: group0 = {count/type, lds_addr, global_addr},
// group1 = {data_size, tensor dims/strides, tile dims}. 2D tile of bf16:
// tile_dim0 = row_elems (contiguous), tile_dim1 = tile_rows.
static __device__ __forceinline__ void tdm_load_tile_bf16(
    unsigned lds_addr, const __bf16* gptr, unsigned tile_rows,
    unsigned row_elems, unsigned tensor_rows) {
  const unsigned long long ga = (unsigned long long)(uintptr_t)gptr;
  u32x4 g0;
  g0.x = 1u;                                            // count=1, user mode
  g0.y = lds_addr;                                      // LDS byte address
  g0.z = (unsigned)ga;                                  // global addr [31:0]
  g0.w = (unsigned)((ga >> 32) & 0x01ffffffu) | (2u << 30);  // [56:32]|type=2
  i32x8 g1;
  g1[0] = (int)(1u << 16);                              // data_size = 2 bytes
  g1[1] = (int)((row_elems & 0xffffu) << 16);           // tensor_dim0 [15:0]
  g1[2] = (int)(((row_elems >> 16) & 0xffffu) |         // tensor_dim0 [31:16]
                ((tensor_rows & 0xffffu) << 16));       // tensor_dim1 [15:0]
  g1[3] = (int)(((tensor_rows >> 16) & 0xffffu) |       // tensor_dim1 [31:16]
                ((row_elems & 0xffffu) << 16));         // tile_dim0
  g1[4] = (int)(tile_rows & 0xffffu);                   // tile_dim1 (dim2=0)
  g1[5] = (int)row_elems;                               // dim0 stride [31:0]
  g1[6] = 0;                                            // stride hi bits
  g1[7] = 0;
  const i32x4 z4 = {0, 0, 0, 0};
#if __clang_major__ >= 23
  const i32x8 z8 = {0, 0, 0, 0, 0, 0, 0, 0};
  __builtin_amdgcn_tensor_load_to_lds(g0, g1, z4, z4, z8, 0);
#else
  __builtin_amdgcn_tensor_load_to_lds(g0, g1, z4, z4, 0);
#endif
}

// --------------------------- weight norm (tiny) ----------------------------
__global__ void wnorm_kernel(const float* __restrict__ v,
                             const float* __restrict__ g,
                             __bf16* __restrict__ w) {
  __shared__ float red[256];
  const int h = blockIdx.x;
  const int d = threadIdx.x;
  const float x = v[h * 256 + d];
  red[d] = x * x;
  __syncthreads();
  for (int s = 128; s > 0; s >>= 1) {
    if (d < s) red[d] += red[d + s];
    __syncthreads();
  }
  const float scale = g[h] * rsqrtf(red[0]);
  w[h * 256 + d] = (__bf16)(x * scale);
}

// --------------------------- f32 -> bf16 cast ------------------------------
__global__ void cvt_bf16_kernel(const float* __restrict__ x,
                                __bf16* __restrict__ y, int n) {
  const int i = blockIdx.x * blockDim.x + threadIdx.x;
  if (i < n) y[i] = (__bf16)x[i];
}

// --------------------------- projection ------------------------------------
// H = relu(X @ W^T + bias); X: (nrows x 256) bf16, W: (256 x 256) bf16.
__global__ void proj_kernel(const __bf16* __restrict__ X,
                            const __bf16* __restrict__ W,
                            const float* __restrict__ bias,
                            __bf16* __restrict__ Hout) {
  const int lane = threadIdx.x & 31;
  const int wave = threadIdx.x >> 5;
  const int half = lane >> 4;
  const int l16  = lane & 15;
  const int row0 = blockIdx.x * 128 + wave * 16;

  const __bf16* xrow = X + (size_t)(row0 + l16) * 256;   // A-operand row
  for (int nt = 0; nt < 16; ++nt) {
    const __bf16* wrow = W + (size_t)(nt * 16 + l16) * 256;  // B-operand col
    v8f acc = {};
#pragma unroll
    for (int kk = 0; kk < 8; ++kk) {
      v16bf aop = load_op16(xrow + kk * 32 + half * 8);
      v16bf bop = load_op16(wrow + kk * 32 + half * 8);
      acc = wmma_bf16(aop, bop, acc);
    }
    const float bv = bias[nt * 16 + l16];
#pragma unroll
    for (int r = 0; r < 8; ++r) {
      float val = acc[r] + bv;
      val = val > 0.f ? val : 0.f;
      Hout[(size_t)(row0 + half * 8 + r) * 256 + nt * 16 + l16] = (__bf16)val;
    }
  }
}

// --------------------------- flash attention -------------------------------
// Out[b, i, :] = sum_j softmax_j(tau * Q_i . K_j + maskbias_j) * V[b, j, :]
// Block: 256 threads = 8 waves, 128 queries/block, 32-key tiles streamed.
// K tiles arrive via TDM (double buffered); V tile staged transposed by VALU.
__global__ void flash_kernel(const __bf16* __restrict__ Q,
                             const __bf16* __restrict__ K,
                             const __bf16* __restrict__ V,
                             const int*   __restrict__ kmask,
                             const float* __restrict__ tau_p,
                             float* __restrict__ Out, int L) {
  __shared__ __attribute__((aligned(16))) __bf16 Kbuf[2][32 * 256]; // 2x16KB
  __shared__ __attribute__((aligned(16))) __bf16 Vt[256 * 32];      // [feat][key]
  __shared__ __attribute__((aligned(16))) __bf16 Pt[8][16 * 32];    // per-wave P
  __shared__ float mb[32];

  const int lane = threadIdx.x & 31;
  const int wave = threadIdx.x >> 5;
  const int half = lane >> 4;
  const int l16  = lane & 15;
  const int bb   = blockIdx.y;
  const int q0   = blockIdx.x * 128 + wave * 16;
  const float tau = *tau_p;
  const int ntiles = L / 32;

  const __bf16* qrow = Q + ((size_t)bb * L + q0 + l16) * 256;  // A-operand row
  const __bf16* kbase = K + (size_t)bb * L * 256;

  float m[8], l[8];
  v8f O[16];
#pragma unroll
  for (int r = 0; r < 8; ++r) { m[r] = -3.0e38f; l[r] = 0.f; }
#pragma unroll
  for (int nd = 0; nd < 16; ++nd) O[nd] = {};

  // Prologue: TDM-load first K tile into buffer 0 (wave 0 only; uniform D#).
  if (wave == 0)
    tdm_load_tile_bf16((unsigned)(uintptr_t)&Kbuf[0][0], kbase, 32, 256, L);

  for (int tt = 0; tt < ntiles; ++tt) {
    const int t0  = tt * 32;
    const int cur = tt & 1;

    // Stage V tile transposed: Vt[feat][key] (overlaps with in-flight TDM).
    {
      const __bf16* src = V + ((size_t)bb * L + t0) * 256;
      const int key = threadIdx.x >> 3;
      const int fb  = (threadIdx.x & 7) * 32;
      union { u32x4 q; __bf16 e[8]; } tmp;
#pragma unroll
      for (int c = 0; c < 4; ++c) {
        tmp.q = *(const u32x4*)(src + (size_t)key * 256 + fb + c * 8);
#pragma unroll
        for (int j = 0; j < 8; ++j) Vt[(fb + c * 8 + j) * 32 + key] = tmp.e[j];
      }
    }
    if (threadIdx.x < 32)
      mb[threadIdx.x] =
          (kmask[(size_t)bb * L + t0 + threadIdx.x] > 0) ? 0.f : -1e9f;

    // Issue next K tile, then guarantee the current one has landed.
    // TDM ops from one wave complete in order: after issuing tile tt+1 there
    // are <=2 outstanding, so tensorcnt<=1 proves tile tt is resident.
    if (wave == 0) {
      if (tt + 1 < ntiles) {
        tdm_load_tile_bf16((unsigned)(uintptr_t)&Kbuf[cur ^ 1][0],
                           kbase + (size_t)(t0 + 32) * 256, 32, 256, L);
        __builtin_amdgcn_s_wait_tensorcnt(1);
      } else {
        __builtin_amdgcn_s_wait_tensorcnt(0);
      }
    }
    __syncthreads();  // K tile (TDM), V tile, mask all visible block-wide

    const __bf16* Kt = &Kbuf[cur][0];

    // S (16 x 32) = Q_strip . Kt^T, two 16-col halves, K-dim = 256.
    v8f S[2];
#pragma unroll
    for (int n = 0; n < 2; ++n) {
      v8f acc = {};
      const __bf16* krow = Kt + (size_t)(n * 16 + l16) * 256;  // B col = key
#pragma unroll
      for (int kk = 0; kk < 8; ++kk) {
        v16bf aop = load_op16(qrow + kk * 32 + half * 8);
        v16bf bop = load_op16(krow + kk * 32 + half * 8);
        acc = wmma_bf16(aop, bop, acc);
      }
      S[n] = acc;
    }
#pragma unroll
    for (int n = 0; n < 2; ++n) {
      const float mbias = mb[n * 16 + l16];
#pragma unroll
      for (int r = 0; r < 8; ++r) S[n][r] = S[n][r] * tau + mbias;
    }

    // Online softmax. C layout: element r <-> row (half*8 + r), col l16;
    // a row lives entirely inside one 16-lane half -> xor-shuffle reduce.
    float mnew[8], f[8];
#pragma unroll
    for (int r = 0; r < 8; ++r) {
      float mx = fmaxf(S[0][r], S[1][r]);
#pragma unroll
      for (int sh = 1; sh < 16; sh <<= 1) mx = fmaxf(mx, __shfl_xor(mx, sh, 32));
      mnew[r] = fmaxf(m[r], mx);
      f[r] = __expf(m[r] - mnew[r]);
      m[r] = mnew[r];
    }

    float rs[8];
#pragma unroll
    for (int r = 0; r < 8; ++r) rs[r] = 0.f;
#pragma unroll
    for (int n = 0; n < 2; ++n) {
#pragma unroll
      for (int r = 0; r < 8; ++r) {
        const float p = __expf(S[n][r] - mnew[r]);
        rs[r] += p;
        Pt[wave][(half * 8 + r) * 32 + n * 16 + l16] = (__bf16)p;  // row-major
      }
    }
#pragma unroll
    for (int r = 0; r < 8; ++r) {
      float s = rs[r];
#pragma unroll
      for (int sh = 1; sh < 16; sh <<= 1) s += __shfl_xor(s, sh, 32);
      l[r] = l[r] * f[r] + s;
    }

    // Wave-local LDS RAW (C-layout P -> A-layout reload): drain DS counter.
    asm volatile("s_wait_dscnt 0" ::: "memory");

    // O = O*f + P @ V_tile  (K-dim = 32 keys, 16 feature tiles).
    const __bf16* prow = &Pt[wave][l16 * 32];  // A-operand row = query row
#pragma unroll
    for (int nd = 0; nd < 16; ++nd) {
#pragma unroll
      for (int r = 0; r < 8; ++r) O[nd][r] *= f[r];
      v16bf aop = load_op16(prow + half * 8);
      const __bf16* vcol = Vt + (size_t)(nd * 16 + l16) * 32;  // B col = feat
      v16bf bop = load_op16(vcol + half * 8);
      O[nd] = wmma_bf16(aop, bop, O[nd]);
    }

    __syncthreads();  // all readers done before Vt/Kbuf[cur] are overwritten
  }

#pragma unroll
  for (int nd = 0; nd < 16; ++nd) {
#pragma unroll
    for (int r = 0; r < 8; ++r) {
      Out[((size_t)bb * L + q0 + half * 8 + r) * 256 + nd * 16 + l16] =
          O[nd][r] / l[r];
    }
  }
}

// ---------------------------------------------------------------------------
extern "C" void kernel_launch(void* const* d_in, const int* in_sizes, int n_in,
                              void* d_out, int out_size, void* d_ws,
                              size_t ws_size, hipStream_t stream) {
  (void)in_sizes; (void)n_in; (void)out_size; (void)ws_size;

  const float* a      = (const float*)d_in[0];
  const float* b      = (const float*)d_in[1];
  const int*   mask_a = (const int*)d_in[2];
  const int*   mask_b = (const int*)d_in[3];
  const float* a_v    = (const float*)d_in[4];
  const float* a_g    = (const float*)d_in[5];
  const float* a_bias = (const float*)d_in[6];
  const float* b_v    = (const float*)d_in[7];
  const float* b_g    = (const float*)d_in[8];
  const float* b_bias = (const float*)d_in[9];
  const float* tau    = (const float*)d_in[10];

  const int B = 8, L = 2048, D = 256, H = 256;
  const size_t NR = (size_t)B * L;  // 16384 rows

  char* ws = (char*)d_ws;
  __bf16* aB = (__bf16*)ws; ws += NR * D * sizeof(__bf16);
  __bf16* bB = (__bf16*)ws; ws += NR * D * sizeof(__bf16);
  __bf16* wA = (__bf16*)ws; ws += (size_t)H * D * sizeof(__bf16);
  __bf16* wB = (__bf16*)ws; ws += (size_t)H * D * sizeof(__bf16);
  __bf16* ha = (__bf16*)ws; ws += NR * H * sizeof(__bf16);
  __bf16* hb = (__bf16*)ws; ws += NR * H * sizeof(__bf16);

  float* out_a = (float*)d_out;
  float* out_b = out_a + NR * D;

  wnorm_kernel<<<H, D, 0, stream>>>(a_v, a_g, wA);
  wnorm_kernel<<<H, D, 0, stream>>>(b_v, b_g, wB);

  const int n = (int)(NR * D);
  cvt_bf16_kernel<<<(n + 255) / 256, 256, 0, stream>>>(a, aB, n);
  cvt_bf16_kernel<<<(n + 255) / 256, 256, 0, stream>>>(b, bB, n);

  proj_kernel<<<(int)(NR / 128), 256, 0, stream>>>(aB, wA, a_bias, ha);
  proj_kernel<<<(int)(NR / 128), 256, 0, stream>>>(bB, wB, b_bias, hb);

  dim3 fg(L / 128, B);
  flash_kernel<<<fg, 256, 0, stream>>>(ha, hb, bB, mask_b, tau, out_a, L);
  flash_kernel<<<fg, 256, 0, stream>>>(hb, ha, aB, mask_a, tau, out_b, L);
}
